// TriangleAttentionStartingNode_17222818857191
// MI455X (gfx1250) — matrix operand order repaired and checked by reading, not compiled, not run on hardware
//
#include <hip/hip_runtime.h>
#include <stdint.h>

#define N_RES 256
#define DIM   128
#define NH    4
#define NC    32
#define WCAT_ROWS 576   // 512 qkvg + 4 bias + pad to multiple of 64

typedef __attribute__((ext_vector_type(16))) __bf16 v16bf;
typedef __attribute__((ext_vector_type(8)))  float  v8f;

union Frag16 { v16bf v; uint16_t u[16]; uint4 q[2]; };
union Acc8   { v8f   v; float    f[8]; };

__device__ __forceinline__ uint16_t f2bf(float f) {
  uint32_t u = __float_as_uint(f);
  uint32_t r = u + 0x7FFFu + ((u >> 16) & 1u);   // round-to-nearest-even
  return (uint16_t)(r >> 16);
}
__device__ __forceinline__ float bf2f(uint16_t s) {
  return __uint_as_float(((uint32_t)s) << 16);
}

__device__ __forceinline__ v8f wmma_bf16(const Frag16& a, const Frag16& b, v8f c) {
  return __builtin_amdgcn_wmma_f32_16x16x32_bf16(
      false, a.v, false, b.v, (short)0, c, false, false);
}

// A fragment: 16x32 bf16 tile from row-major [*, ld] at (m0, k0).
// Lane L holds row m0+(L&15); halves 0-7 = K {koff..koff+8}, 8-15 = K {koff+16..koff+24},
// koff = (L<16)?0:8  (per CDNA5 ISA 16-bit A layout).
__device__ __forceinline__ Frag16 load_a(const uint16_t* base, int ld, int m0, int k0, int lane) {
  Frag16 f;
  int row  = m0 + (lane & 15);
  int koff = (lane < 16) ? 0 : 8;
  const uint16_t* p = base + (size_t)row * ld + k0 + koff;
  f.q[0] = *(const uint4*)(p);
  f.q[1] = *(const uint4*)(p + 16);
  return f;
}

// B fragment: 32x16 bf16 tile where B[k][n] = W[n][k], W row-major [nout, ldk].
// Lane L holds column n0+(L&15); halves 0-7 = K {koff..koff+8}, 8-15 = K {koff+16..koff+24},
// koff = (L<16)?0:8 (same K order as load_a).
__device__ __forceinline__ Frag16 load_b_wt(const uint16_t* base, int ld, int n0, int k0, int lane) {
  Frag16 f;
  int col  = n0 + (lane & 15);
  int koff = (lane < 16) ? 0 : 8;
  const uint16_t* p = base + (size_t)col * ld + k0 + koff;
  f.q[0] = *(const uint4*)(p);
  f.q[1] = *(const uint4*)(p + 16);
  return f;
}

// ---------------- Kernel 1: LayerNorm -> bf16 x ----------------
__global__ void ln_kernel(const float* __restrict__ pair,
                          const float* __restrict__ gamma,
                          const float* __restrict__ beta,
                          uint16_t* __restrict__ xbf) {
  int wave = threadIdx.x >> 5;
  int lane = threadIdx.x & 31;
  int row  = blockIdx.x * 8 + wave;                 // 0..65535
  const float* p = pair + (size_t)row * DIM + lane * 4;
  float4 x = *(const float4*)p;
  float s  = x.x + x.y + x.z + x.w;
  float ss = x.x*x.x + x.y*x.y + x.z*x.z + x.w*x.w;
  for (int off = 1; off < 32; off <<= 1) {
    s  += __shfl_xor(s,  off, 32);
    ss += __shfl_xor(ss, off, 32);
  }
  float mu  = s * (1.0f / DIM);
  float var = ss * (1.0f / DIM) - mu * mu;
  float rs  = __frsqrt_rn(var + 1e-5f);
  float4 g = *(const float4*)(gamma + lane * 4);
  float4 b = *(const float4*)(beta  + lane * 4);
  uint16_t o0 = f2bf((x.x - mu) * rs * g.x + b.x);
  uint16_t o1 = f2bf((x.y - mu) * rs * g.y + b.y);
  uint16_t o2 = f2bf((x.z - mu) * rs * g.z + b.z);
  uint16_t o3 = f2bf((x.w - mu) * rs * g.w + b.w);
  uint2 st;
  st.x = (uint32_t)o0 | ((uint32_t)o1 << 16);
  st.y = (uint32_t)o2 | ((uint32_t)o3 << 16);
  *(uint2*)(xbf + (size_t)row * DIM + lane * 4) = st;
}

// ---------------- Kernel 2: weight pack/convert to bf16 ----------------
// wcat rows: 0-127 Wq, 128-255 Wk, 256-383 Wv, 384-511 Wg, 512-515 Wb, 516-575 zero.
__global__ void wconv_kernel(const float* __restrict__ Wq, const float* __restrict__ Wk,
                             const float* __restrict__ Wv, const float* __restrict__ Wb,
                             const float* __restrict__ Wg, const float* __restrict__ Wo,
                             uint16_t* __restrict__ wcat, uint16_t* __restrict__ wobf) {
  int idx = blockIdx.x * blockDim.x + threadIdx.x;
  const int total1 = WCAT_ROWS * 128;
  if (idx < total1) {
    int r = idx >> 7, k = idx & 127;
    float val = 0.0f;
    if      (r < 128) val = Wq[(size_t)r * 128 + k];
    else if (r < 256) val = Wk[(size_t)(r - 128) * 128 + k];
    else if (r < 384) val = Wv[(size_t)(r - 256) * 128 + k];
    else if (r < 512) val = Wg[(size_t)(r - 384) * 128 + k];
    else if (r < 516) val = Wb[(size_t)(r - 512) * 128 + k];
    wcat[idx] = f2bf(val);
  } else if (idx < total1 + 128 * 128) {
    int j = idx - total1;
    wobf[j] = f2bf(Wo[j]);
  }
}

// ---------------- Kernel 3: fused q/k/v/g/b projection GEMM ----------------
// [65536 x 128] x [128 x 576]^T.  Each wave: 16x64 tile (one A fragment feeds
// 4 WMMAs per K-step). 4096 M-tiles x 9 N-groups = 36864 waves.
__global__ void proj_kernel(const uint16_t* __restrict__ xbf, const uint16_t* __restrict__ wcat,
                            uint16_t* __restrict__ qb, uint16_t* __restrict__ kb,
                            uint16_t* __restrict__ vb, uint16_t* __restrict__ gb,
                            float* __restrict__ bb) {
  int lane = threadIdx.x & 31;
  int wave = threadIdx.x >> 5;
  int tile = blockIdx.x * 8 + wave;
  int mt = tile / 9, ng = tile % 9;
  int m0 = mt * 16, n0 = ng * 64;
  Acc8 acc[4];
  for (int t = 0; t < 4; ++t)
    for (int r = 0; r < 8; ++r) acc[t].f[r] = 0.0f;
  for (int k0 = 0; k0 < 128; k0 += 32) {
    Frag16 a = load_a(xbf, DIM, m0, k0, lane);
    for (int t = 0; t < 4; ++t) {
      Frag16 b = load_b_wt(wcat, DIM, n0 + 16 * t, k0, lane);
      acc[t].v = wmma_bf16(a, b, acc[t].v);
    }
  }
  const float qscale = 1.0f / (float)(NC * NC);
  int mbase = m0 + ((lane < 16) ? 0 : 8);
  for (int t = 0; t < 4; ++t) {
    int n = n0 + 16 * t + (lane & 15);
    for (int r = 0; r < 8; ++r) {
      int m = mbase + r;
      int i = m >> 8, j = m & 255;
      float val = acc[t].f[r];
      if (n < 128) {
        int h = n >> 5, c = n & 31;
        qb[((size_t)(i * NH + h) * N_RES + j) * NC + c] = f2bf(val * qscale);
      } else if (n < 256) {
        int nn = n - 128; int h = nn >> 5, c = nn & 31;
        kb[((size_t)(i * NH + h) * N_RES + j) * NC + c] = f2bf(val);
      } else if (n < 384) {
        int nn = n - 256; int h = nn >> 5, c = nn & 31;
        vb[((size_t)(i * NH + h) * N_RES + j) * NC + c] = f2bf(val);
      } else if (n < 512) {
        gb[(size_t)m * DIM + (n - 384)] = f2bf(val);
      } else if (n < 516) {
        int h = n - 512;
        bb[((size_t)h * N_RES + i) * N_RES + j] = val;
      }
    }
  }
}

// ---------------- Kernel 4: attention per (i, h) ----------------
// Block = 128 threads (4 waves); each wave does 4 tiles of 16 q-rows.
// LDS: K[256][32] bf16 + V^T[32][256] bf16 + probs[4 waves][16][256] bf16 = 64 KB.
__global__ void attn_kernel(const uint16_t* __restrict__ qb, const uint16_t* __restrict__ kb,
                            const uint16_t* __restrict__ vb, const uint16_t* __restrict__ gbuf,
                            const float* __restrict__ bb, uint16_t* __restrict__ gated) {
  extern __shared__ uint16_t smem[];
  uint16_t* k_s   = smem;                       // 256*32
  uint16_t* vT_s  = smem + 256 * 32;            // 32*256
  uint16_t* probs = vT_s + 32 * 256;            // 4*16*256
  int i = blockIdx.x, h = blockIdx.y;
  int tid = threadIdx.x, lane = tid & 31, wave = tid >> 5;
  size_t base = (size_t)(i * NH + h) * N_RES * NC;

  // Stage K (row-major) and V (transposed) into LDS.
#if defined(__gfx1250__)
  for (int jj = tid; jj < N_RES; jj += 128) {
    const uint16_t* src = kb + base + (size_t)jj * NC;
    uint16_t* dst = k_s + jj * NC;
    // CDNA5 async memory->LDS copy (VFLAT async, tracked by ASYNCcnt).
    // vdst carries the wave-relative LDS byte offset (= low 32 bits of the
    // generic pointer for the LDS aperture), vaddr is the 64-bit global addr.
    for (int c = 0; c < 4; ++c) {
      uint32_t ldsoff = (uint32_t)(uintptr_t)(dst + 8 * c);
      asm volatile("global_load_async_to_lds_b128 %0, %1, off"
                   :: "v"(ldsoff), "v"(src + 8 * c) : "memory");
    }
    const uint16_t* vr = vb + base + (size_t)jj * NC;
    for (int c = 0; c < NC; ++c) vT_s[c * N_RES + jj] = vr[c];
  }
  asm volatile("s_wait_asynccnt 0x0" ::: "memory");
#else
  for (int jj = tid; jj < N_RES; jj += 128) {
    const uint4* src = (const uint4*)(kb + base + (size_t)jj * NC);
    uint4* dst = (uint4*)(k_s + jj * NC);
    dst[0] = src[0]; dst[1] = src[1]; dst[2] = src[2]; dst[3] = src[3];
    const uint16_t* vr = vb + base + (size_t)jj * NC;
    for (int c = 0; c < NC; ++c) vT_s[c * N_RES + jj] = vr[c];
  }
#endif
  __syncthreads();

  uint16_t* pw = probs + wave * 16 * N_RES;
  const float* bh = bb + (size_t)h * N_RES * N_RES;
  int mloc = (lane < 16) ? 0 : 8;
  int ncol = lane & 15;

  for (int sub = 0; sub < 4; ++sub) {
    int m0 = (wave * 4 + sub) * 16;              // q-row tile base
    Frag16 aq = load_a(qb + base, NC, m0, 0, lane);

    // logits = (q*scale) . K^T + bias  (bias folded into C operand)
    Acc8 acc[16];
    for (int t = 0; t < 16; ++t) {
      for (int r = 0; r < 8; ++r)
        acc[t].f[r] = bh[(size_t)(m0 + mloc + r) * N_RES + t * 16 + ncol];
      Frag16 bk = load_b_wt(k_s, NC, t * 16, 0, lane);
      acc[t].v = wmma_bf16(aq, bk, acc[t].v);
    }

    // softmax: each logit row lives in one 16-lane half -> xor reduce 1,2,4,8
    for (int r = 0; r < 8; ++r) {
      float mx = -3.0e38f;
      for (int t = 0; t < 16; ++t) mx = fmaxf(mx, acc[t].f[r]);
      for (int off = 1; off < 16; off <<= 1) mx = fmaxf(mx, __shfl_xor(mx, off, 32));
      float sum = 0.0f;
      for (int t = 0; t < 16; ++t) {
        float e = __expf(acc[t].f[r] - mx);
        acc[t].f[r] = e;
        sum += e;
      }
      for (int off = 1; off < 16; off <<= 1) sum += __shfl_xor(sum, off, 32);
      float inv = 1.0f / sum;
      for (int t = 0; t < 16; ++t) acc[t].f[r] *= inv;
    }

    // probs -> LDS (bf16, row-major 16x256) for re-fragmentation as A
    for (int t = 0; t < 16; ++t)
      for (int r = 0; r < 8; ++r)
        pw[(mloc + r) * N_RES + t * 16 + ncol] = f2bf(acc[t].f[r]);

    // out = probs . V   (K=256 -> 8 WMMA steps; N=32 -> 2 accumulators)
    Acc8 accO[2];
    for (int tt = 0; tt < 2; ++tt)
      for (int r = 0; r < 8; ++r) accO[tt].f[r] = 0.0f;
    for (int s = 0; s < 8; ++s) {
      Frag16 ap = load_a(pw, N_RES, 0, s * 32, lane);
      for (int tt = 0; tt < 2; ++tt) {
        Frag16 bv = load_b_wt(vT_s, N_RES, tt * 16, s * 32, lane);
        accO[tt].v = wmma_bf16(ap, bv, accO[tt].v);
      }
    }

    // gate and store bf16 gated activations, layout [i*N+jq, h*32+c]
    for (int tt = 0; tt < 2; ++tt)
      for (int r = 0; r < 8; ++r) {
        int jq = m0 + mloc + r;
        int c  = tt * 16 + ncol;
        size_t gi = (size_t)(i * N_RES + jq) * DIM + h * NC + c;
        float gv = bf2f(gbuf[gi]);
        gated[gi] = f2bf(gv * accO[tt].f[r]);
      }
  }
}

// ---------------- Kernel 5: output GEMM gated . Wo^T + bo ----------------
// Each wave: 16x64 tile. 4096 M-tiles x 2 N-groups = 8192 waves.
__global__ void outgemm_kernel(const uint16_t* __restrict__ gated,
                               const uint16_t* __restrict__ wobf,
                               const float* __restrict__ bo,
                               float* __restrict__ out) {
  int lane = threadIdx.x & 31;
  int wave = threadIdx.x >> 5;
  int tile = blockIdx.x * 8 + wave;
  int mt = tile >> 1, ng = tile & 1;
  int m0 = mt * 16, n0 = ng * 64;
  Acc8 acc[4];
  for (int t = 0; t < 4; ++t) {
    float bv = bo[n0 + 16 * t + (lane & 15)];
    for (int r = 0; r < 8; ++r) acc[t].f[r] = bv;
  }
  for (int k0 = 0; k0 < 128; k0 += 32) {
    Frag16 a = load_a(gated, DIM, m0, k0, lane);
    for (int t = 0; t < 4; ++t) {
      Frag16 b = load_b_wt(wobf, DIM, n0 + 16 * t, k0, lane);
      acc[t].v = wmma_bf16(a, b, acc[t].v);
    }
  }
  int mbase = m0 + ((lane < 16) ? 0 : 8);
  for (int t = 0; t < 4; ++t) {
    int n = n0 + 16 * t + (lane & 15);
    for (int r = 0; r < 8; ++r)
      out[(size_t)(mbase + r) * DIM + n] = acc[t].f[r];
  }
}

extern "C" void kernel_launch(void* const* d_in, const int* in_sizes, int n_in,
                              void* d_out, int out_size, void* d_ws, size_t ws_size,
                              hipStream_t stream) {
  (void)in_sizes; (void)n_in; (void)out_size; (void)ws_size;
  const float* pair  = (const float*)d_in[0];
  const float* gamma = (const float*)d_in[1];
  const float* beta  = (const float*)d_in[2];
  const float* Wq    = (const float*)d_in[3];
  const float* Wk    = (const float*)d_in[4];
  const float* Wv    = (const float*)d_in[5];
  const float* Wb    = (const float*)d_in[6];
  const float* Wg    = (const float*)d_in[7];
  const float* Wo    = (const float*)d_in[8];
  const float* bo    = (const float*)d_in[9];
  float* out = (float*)d_out;

  char* ws = (char*)d_ws;
  size_t off = 0;
  auto alloc = [&](size_t bytes) -> char* {
    char* p = ws + off;
    off += (bytes + 255) & ~(size_t)255;
    return p;
  };
  const size_t M = (size_t)N_RES * N_RES;         // 65536
  uint16_t* xbf   = (uint16_t*)alloc(M * DIM * 2);
  uint16_t* wcat  = (uint16_t*)alloc((size_t)WCAT_ROWS * 128 * 2);
  uint16_t* wobf  = (uint16_t*)alloc(128ull * 128 * 2);
  uint16_t* qb    = (uint16_t*)alloc((size_t)N_RES * NH * N_RES * NC * 2);
  uint16_t* kb    = (uint16_t*)alloc((size_t)N_RES * NH * N_RES * NC * 2);
  uint16_t* vb    = (uint16_t*)alloc((size_t)N_RES * NH * N_RES * NC * 2);
  uint16_t* gb    = (uint16_t*)alloc(M * DIM * 2);
  float*    bbuf  = (float*)   alloc((size_t)NH * N_RES * N_RES * 4);
  uint16_t* gated = (uint16_t*)alloc(M * DIM * 2);

  ln_kernel<<<8192, 256, 0, stream>>>(pair, gamma, beta, xbf);
  wconv_kernel<<<(WCAT_ROWS * 128 + 128 * 128 + 255) / 256, 256, 0, stream>>>(
      Wq, Wk, Wv, Wb, Wg, Wo, wcat, wobf);
  proj_kernel<<<4608, 256, 0, stream>>>(xbf, wcat, qb, kb, vb, gb, bbuf);
  dim3 agrid(N_RES, NH);
  attn_kernel<<<agrid, 128, 64 * 1024, stream>>>(qb, kb, vb, gb, bbuf, gated);
  outgemm_kernel<<<1024, 256, 0, stream>>>(gated, wobf, bo, out);
}
